// SelfAttention_28226525069887
// MI455X (gfx1250) — matrix operand order repaired and checked by reading, not compiled, not run on hardware
//
#include <hip/hip_runtime.h>

// MI455X / gfx1250: wave32, WMMA 16x16x32 bf16 -> f32.
// SelfAttention with fused lora adapters: B=2, N=2048, D=512, H=8, DK=DV=64.

#define DMODEL 512
#define NSEQ   2048
#define NBATCH 2
#define NHEAD  8
#define DHEAD  64

typedef __attribute__((ext_vector_type(16))) __bf16 v16bf;
typedef __attribute__((ext_vector_type(8)))  __bf16 v8bf;
typedef __attribute__((ext_vector_type(4)))  __bf16 v4bf;
typedef __attribute__((ext_vector_type(8)))  float  v8f;

static __device__ __forceinline__ v16bf frag_cat(v8bf lo, v8bf hi) {
  v16bf a;
#pragma unroll
  for (int i = 0; i < 8; ++i) { a[i] = lo[i]; a[i + 8] = hi[i]; }
  return a;
}

// A fragment: 16(M) x 32(K), row-major source [row][k], p = &row0,k0-base.
// ISA 7.12.2 16-bit A layout: row = lane&15, half=lane>>4,
//   pairs 0..3 -> K = half*8 + 0..7 ; pairs 4..7 -> K = 16 + half*8 + 0..7.
static __device__ __forceinline__ v16bf load_a_frag(const __bf16* p, int ld, int k0) {
  const int lane = threadIdx.x & 31;
  const int half = lane >> 4, r = lane & 15;
  const __bf16* q = p + (size_t)r * ld + k0 + half * 8;
  return frag_cat(*(const v8bf*)q, *(const v8bf*)(q + 16));
}

// B fragment: 32(K) x 16(N) from an N-major tile (tile[n][k], leading dim ld).
// N = lane&15 (+n0); K = half*8 + 0..7 and 16 + half*8 + 0..7 (matches A layout).
static __device__ __forceinline__ v16bf load_b_frag(const __bf16* p, int ld, int n0, int k0) {
  const int lane = threadIdx.x & 31;
  const int half = lane >> 4, r = lane & 15;
  const __bf16* q = p + (size_t)(n0 + r) * ld + k0 + half * 8;
  return frag_cat(*(const v8bf*)q, *(const v8bf*)(q + 16));
}

static __device__ __forceinline__ v8f wmma_bf16(v16bf a, v16bf b, v8f c) {
  return __builtin_amdgcn_wmma_f32_16x16x32_bf16(false, a, false, b, (short)0, c,
                                                 false, false);
}

// ---------------- prep kernels (one-time, cheap) ----------------

// WT[n][k] = W[k][n] + Wl[k][n]  (bf16, N-major so GEMM B staging is contiguous)
__global__ void fuse_wT_kernel(const float* __restrict__ W,
                               const float* __restrict__ Wl,
                               __bf16* __restrict__ WT) {
  int idx = blockIdx.x * 256 + threadIdx.x;
  if (idx >= DMODEL * DMODEL) return;
  int n = idx & (DMODEL - 1), k = idx >> 9;
  float v = W[(size_t)k * DMODEL + n];
  if (Wl) v += Wl[(size_t)k * DMODEL + n];
  WT[(size_t)n * DMODEL + k] = (__bf16)v;
}

__global__ void fuse_bias_kernel(const float* __restrict__ b0,
                                 const float* __restrict__ b1,
                                 float* __restrict__ out) {
  int i = blockIdx.x * 256 + threadIdx.x;
  if (i < DMODEL) out[i] = b0[i] + (b1 ? b1[i] : 0.f);
}

__global__ void cvt_x_kernel(const float* __restrict__ x, __bf16* __restrict__ xb,
                             int n4) {
  int i = blockIdx.x * 256 + threadIdx.x;
  if (i >= n4) return;
  float4 v = ((const float4*)x)[i];
  v4bf o;
  o[0] = (__bf16)v.x; o[1] = (__bf16)v.y; o[2] = (__bf16)v.z; o[3] = (__bf16)v.w;
  ((v4bf*)xb)[i] = o;
}

// ---------------- fused QKV projection GEMM ----------------
// C[4096,512] = Xb[4096,512] @ (W+Wl)[512,512] + bias.
// Block tile 128(M) x 256(N); 8 waves in 2x4 -> 64x64 per wave = 16 WMMA/k-step.
// mode = blockIdx.z: 0 -> Q*(1/8) into [b,h,tok,d]; 1 -> K into [b,h,tok,d];
//                    2 -> V into [b,h,d,tok] (pre-transposed for P@V B-frags).
__global__ __launch_bounds__(256) void gemm_qkv_kernel(
    const __bf16* __restrict__ X, const __bf16* __restrict__ WT3,
    const float* __restrict__ bias3,
    __bf16* __restrict__ Qo, __bf16* __restrict__ Ko, __bf16* __restrict__ Vo) {
  __shared__ __bf16 At[128 * 32];   // [m][k]
  __shared__ __bf16 Bt[256 * 32];   // [n][k]
  const int mode = blockIdx.z;
  const __bf16* WT = WT3 + (size_t)mode * DMODEL * DMODEL;
  const float* bias = bias3 + mode * DMODEL;
  const int m0 = blockIdx.x * 128, n0 = blockIdx.y * 256;
  const int wid = threadIdx.x >> 5, lane = threadIdx.x & 31;
  const int half = lane >> 4, r = lane & 15;
  const int wm = (wid >> 2) * 64, wn = (wid & 3) * 64;

  v8f acc[4][4];
#pragma unroll
  for (int i = 0; i < 4; ++i)
#pragma unroll
    for (int j = 0; j < 4; ++j) { v8f z = {}; acc[i][j] = z; }

  for (int kc = 0; kc < DMODEL; kc += 32) {
    __syncthreads();
    {  // straight-line staging: A = 512 b128 chunks, B = 1024 b128 chunks
      const int c0 = threadIdx.x;
#pragma unroll
      for (int s = 0; s < 2; ++s) {
        int c = c0 + s * 256, row = c >> 2, cc = c & 3;
        ((uint4*)At)[c] = *(const uint4*)(X + (size_t)(m0 + row) * DMODEL + kc + cc * 8);
      }
#pragma unroll
      for (int s = 0; s < 4; ++s) {
        int c = c0 + s * 256, row = c >> 2, cc = c & 3;
        ((uint4*)Bt)[c] = *(const uint4*)(WT + (size_t)(n0 + row) * DMODEL + kc + cc * 8);
      }
    }
    __syncthreads();
    v16bf a[4], b[4];
#pragma unroll
    for (int i = 0; i < 4; ++i) a[i] = load_a_frag(At + (wm + i * 16) * 32, 32, 0);
#pragma unroll
    for (int j = 0; j < 4; ++j) b[j] = load_b_frag(Bt, 32, wn + j * 16, 0);
#pragma unroll
    for (int i = 0; i < 4; ++i)
#pragma unroll
      for (int j = 0; j < 4; ++j) acc[i][j] = wmma_bf16(a[i], b[j], acc[i][j]);
  }

#pragma unroll
  for (int i = 0; i < 4; ++i)
#pragma unroll
    for (int j = 0; j < 4; ++j)
#pragma unroll
      for (int g = 0; g < 8; ++g) {
        int m = m0 + wm + i * 16 + g + half * 8;   // C layout: row = vgpr + 8*half
        int n = n0 + wn + j * 16 + r;              //           col = lane&15
        float v = acc[i][j][g] + bias[n];
        int bb = m >> 11, tok = m & (NSEQ - 1);
        int h = n >> 6, d = n & (DHEAD - 1);
        size_t bh = (size_t)bb * NHEAD + h;
        if (mode == 0)
          Qo[(bh * NSEQ + tok) * DHEAD + d] = (__bf16)(0.125f * v);
        else if (mode == 1)
          Ko[(bh * NSEQ + tok) * DHEAD + d] = (__bf16)v;
        else
          Vo[(bh * DHEAD + d) * NSEQ + tok] = (__bf16)v;
      }
}

// ---------------- flash attention ----------------
// block: one (b,h), 128 queries; 8 waves x 16-query strips; 64-key chunks.
__global__ __launch_bounds__(256) void attn_kernel(
    const __bf16* __restrict__ Q, const __bf16* __restrict__ K,
    const __bf16* __restrict__ VT, __bf16* __restrict__ Aout) {
  __shared__ __bf16 Kt[64 * 64];      // [key][d]   : B-frags for Q@K^T (N=key)
  __shared__ __bf16 Vt[64 * 64];      // [d][key]   : B-frags for P@V   (N=d)
  __shared__ __bf16 Pt[8 * 16 * 64];  // per-wave P re-layout C->A

  const int bh = blockIdx.y;                 // b*8 + h
  const int q0 = blockIdx.x * 128;
  const int wid = threadIdx.x >> 5, lane = threadIdx.x & 31;
  const int half = lane >> 4, r = lane & 15;

  const __bf16* qbase = Q + ((size_t)bh * NSEQ + q0 + wid * 16) * DHEAD;
  const v16bf qa0 = load_a_frag(qbase, DHEAD, 0);
  const v16bf qa1 = load_a_frag(qbase, DHEAD, 32);
  const __bf16* kbase = K + (size_t)bh * NSEQ * DHEAD;
  const __bf16* vbase = VT + (size_t)bh * DHEAD * NSEQ;
  __bf16* pw = Pt + wid * (16 * 64);

  v8f acc[4];
#pragma unroll
  for (int t = 0; t < 4; ++t) { v8f z = {}; acc[t] = z; }
  float mrow[8], lrow[8];
#pragma unroll
  for (int g = 0; g < 8; ++g) { mrow[g] = -1e30f; lrow[g] = 0.f; }

  for (int kc = 0; kc < NSEQ; kc += 64) {
    {  // stage K chunk (64 keys x 64 d, contiguous 8KB) + V^T chunk (64 d x 64 keys)
      const int c0 = threadIdx.x;
#pragma unroll
      for (int s = 0; s < 2; ++s) {
        int c = c0 + s * 256;
        ((uint4*)Kt)[c] = ((const uint4*)(kbase + (size_t)kc * DHEAD))[c];
      }
#pragma unroll
      for (int s = 0; s < 2; ++s) {
        int c = c0 + s * 256, row = c >> 3, cc = c & 7;
        ((uint4*)Vt)[c] = *(const uint4*)(vbase + (size_t)row * NSEQ + kc + cc * 8);
      }
      if (kc + 64 < NSEQ) {  // gfx1250 global_prefetch_b8 for the next chunk
        __builtin_prefetch(kbase + (size_t)(kc + 64) * DHEAD + c0 * 16, 0, 3);
        __builtin_prefetch(vbase + (size_t)(c0 >> 2) * NSEQ + kc + 64, 0, 3);
      }
    }
    __syncthreads();

    // S = (Q*scale) @ K^T : four 16x16 tiles over 64 keys, K-dim = DHEAD in 2 steps
    v8f s[4];
#pragma unroll
    for (int j = 0; j < 4; ++j) {
      v8f z = {};
      z = wmma_bf16(qa0, load_b_frag(Kt, 64, j * 16, 0), z);
      z = wmma_bf16(qa1, load_b_frag(Kt, 64, j * 16, 32), z);
      s[j] = z;
    }

    // online softmax; rows are spread over the 16 lanes of each half-group
#pragma unroll
    for (int g = 0; g < 8; ++g) {
      float cmax = fmaxf(fmaxf(s[0][g], s[1][g]), fmaxf(s[2][g], s[3][g]));
#pragma unroll
      for (int mk = 8; mk >= 1; mk >>= 1) cmax = fmaxf(cmax, __shfl_xor(cmax, mk, 16));
      float mn = fmaxf(mrow[g], cmax);
      float alpha = __expf(mrow[g] - mn);
      float p0 = __expf(s[0][g] - mn);
      float p1 = __expf(s[1][g] - mn);
      float p2 = __expf(s[2][g] - mn);
      float p3 = __expf(s[3][g] - mn);
      float rs = (p0 + p1) + (p2 + p3);
#pragma unroll
      for (int mk = 8; mk >= 1; mk >>= 1) rs += __shfl_xor(rs, mk, 16);
      lrow[g] = lrow[g] * alpha + rs;
      mrow[g] = mn;
#pragma unroll
      for (int t = 0; t < 4; ++t) acc[t][g] *= alpha;
      // spill P in [row][key] so it can be re-read as A fragments
      __bf16* pr = pw + (g + half * 8) * 64 + r;
      pr[0] = (__bf16)p0;
      pr[16] = (__bf16)p1;
      pr[32] = (__bf16)p2;
      pr[48] = (__bf16)p3;
    }
    v16bf pf0 = load_a_frag(pw, 64, 0);   // same-wave LDS RAW -> s_wait_dscnt
    v16bf pf1 = load_a_frag(pw, 64, 32);

    // O += P @ V : N = d in four 16-wide tiles, K = 64 keys in 2 steps
#pragma unroll
    for (int t = 0; t < 4; ++t) {
      acc[t] = wmma_bf16(pf0, load_b_frag(Vt, 64, t * 16, 0), acc[t]);
      acc[t] = wmma_bf16(pf1, load_b_frag(Vt, 64, t * 16, 32), acc[t]);
    }
    __syncthreads();
  }

  const int bb = bh >> 3, h = bh & (NHEAD - 1);
#pragma unroll
  for (int t = 0; t < 4; ++t)
#pragma unroll
    for (int g = 0; g < 8; ++g) {
      int tok = q0 + wid * 16 + g + half * 8;
      int d = t * 16 + r;
      Aout[((size_t)bb * NSEQ + tok) * DMODEL + h * DHEAD + d] =
          (__bf16)(acc[t][g] / lrow[g]);
    }
}

// ---------------- output projection ----------------
__global__ __launch_bounds__(256) void gemm_out_kernel(
    const __bf16* __restrict__ X, const __bf16* __restrict__ WT,
    const float* __restrict__ bias, float* __restrict__ Out) {
  __shared__ __bf16 At[128 * 32];
  __shared__ __bf16 Bt[256 * 32];
  const int m0 = blockIdx.x * 128, n0 = blockIdx.y * 256;
  const int wid = threadIdx.x >> 5, lane = threadIdx.x & 31;
  const int half = lane >> 4, r = lane & 15;
  const int wm = (wid >> 2) * 64, wn = (wid & 3) * 64;

  v8f acc[4][4];
#pragma unroll
  for (int i = 0; i < 4; ++i)
#pragma unroll
    for (int j = 0; j < 4; ++j) { v8f z = {}; acc[i][j] = z; }

  for (int kc = 0; kc < DMODEL; kc += 32) {
    __syncthreads();
    {
      const int c0 = threadIdx.x;
#pragma unroll
      for (int s = 0; s < 2; ++s) {
        int c = c0 + s * 256, row = c >> 2, cc = c & 3;
        ((uint4*)At)[c] = *(const uint4*)(X + (size_t)(m0 + row) * DMODEL + kc + cc * 8);
      }
#pragma unroll
      for (int s = 0; s < 4; ++s) {
        int c = c0 + s * 256, row = c >> 2, cc = c & 3;
        ((uint4*)Bt)[c] = *(const uint4*)(WT + (size_t)(n0 + row) * DMODEL + kc + cc * 8);
      }
    }
    __syncthreads();
    v16bf a[4], b[4];
#pragma unroll
    for (int i = 0; i < 4; ++i) a[i] = load_a_frag(At + (wm + i * 16) * 32, 32, 0);
#pragma unroll
    for (int j = 0; j < 4; ++j) b[j] = load_b_frag(Bt, 32, wn + j * 16, 0);
#pragma unroll
    for (int i = 0; i < 4; ++i)
#pragma unroll
      for (int j = 0; j < 4; ++j) acc[i][j] = wmma_bf16(a[i], b[j], acc[i][j]);
  }

#pragma unroll
  for (int i = 0; i < 4; ++i)
#pragma unroll
    for (int j = 0; j < 4; ++j)
#pragma unroll
      for (int g = 0; g < 8; ++g) {
        int m = m0 + wm + i * 16 + g + half * 8;
        int n = n0 + wn + j * 16 + r;
        Out[(size_t)m * DMODEL + n] = acc[i][j][g] + bias[n];
      }
}

// ---------------- host side ----------------
extern "C" void kernel_launch(void* const* d_in, const int* in_sizes, int n_in,
                              void* d_out, int out_size, void* d_ws, size_t ws_size,
                              hipStream_t stream) {
  const float* x   = (const float*)d_in[0];
  const float* Wq  = (const float*)d_in[1];
  const float* bq  = (const float*)d_in[2];
  const float* Wk  = (const float*)d_in[3];
  const float* bk  = (const float*)d_in[4];
  const float* Wv  = (const float*)d_in[5];
  const float* bv  = (const float*)d_in[6];
  const float* Wql = (const float*)d_in[7];
  const float* bql = (const float*)d_in[8];
  const float* Wkl = (const float*)d_in[9];
  const float* bkl = (const float*)d_in[10];
  const float* Wvl = (const float*)d_in[11];
  const float* bvl = (const float*)d_in[12];
  const float* Wo  = (const float*)d_in[13];
  const float* bo  = (const float*)d_in[14];

  char* ws = (char*)d_ws;
  size_t off = 0;
  auto alloc = [&](size_t bytes) {
    void* p = ws + off;
    off = (off + bytes + 255) & ~(size_t)255;
    return p;
  };
  const size_t tokens = (size_t)NBATCH * NSEQ;             // 4096
  __bf16* xb   = (__bf16*)alloc(tokens * DMODEL * 2);      // x in bf16
  __bf16* WT3  = (__bf16*)alloc((size_t)3 * DMODEL * DMODEL * 2);
  __bf16* WoT  = (__bf16*)alloc((size_t)DMODEL * DMODEL * 2);
  float*  b3   = (float*)alloc((size_t)3 * DMODEL * 4);
  __bf16* q    = (__bf16*)alloc(tokens * DMODEL * 2);      // [b,h,tok,d] scaled
  __bf16* k    = (__bf16*)alloc(tokens * DMODEL * 2);      // [b,h,tok,d]
  __bf16* vt   = (__bf16*)alloc(tokens * DMODEL * 2);      // [b,h,d,tok]
  __bf16* attn = (__bf16*)alloc(tokens * DMODEL * 2);      // [tok, H*DV]
  (void)ws_size; (void)n_in; (void)in_sizes; (void)out_size;

  const int wgrid = (DMODEL * DMODEL + 255) / 256;
  fuse_wT_kernel<<<wgrid, 256, 0, stream>>>(Wq, Wql, WT3);
  fuse_wT_kernel<<<wgrid, 256, 0, stream>>>(Wk, Wkl, WT3 + (size_t)DMODEL * DMODEL);
  fuse_wT_kernel<<<wgrid, 256, 0, stream>>>(Wv, Wvl, WT3 + (size_t)2 * DMODEL * DMODEL);
  fuse_wT_kernel<<<wgrid, 256, 0, stream>>>(Wo, nullptr, WoT);
  fuse_bias_kernel<<<2, 256, 0, stream>>>(bq, bql, b3);
  fuse_bias_kernel<<<2, 256, 0, stream>>>(bk, bkl, b3 + DMODEL);
  fuse_bias_kernel<<<2, 256, 0, stream>>>(bv, bvl, b3 + 2 * DMODEL);
  const int n4 = (int)(tokens * DMODEL / 4);
  cvt_x_kernel<<<(n4 + 255) / 256, 256, 0, stream>>>(x, xb, n4);

  gemm_qkv_kernel<<<dim3(tokens / 128, DMODEL / 256, 3), 256, 0, stream>>>(
      xb, WT3, b3, q, k, vt);

  attn_kernel<<<dim3(NSEQ / 128, NBATCH * NHEAD), 256, 0, stream>>>(q, k, vt, attn);

  gemm_out_kernel<<<dim3(tokens / 128, DMODEL / 256), 256, 0, stream>>>(
      attn, WoT, bo, (float*)d_out);
}
